// SageGnNetwork_24670292149153
// MI455X (gfx1250) — compile-verified, matrix operand order checked by reading, not run on hardware
//
#include <hip/hip_runtime.h>
#include <hip/hip_bf16.h>

typedef __attribute__((ext_vector_type(16))) _Float16 v16h;
typedef __attribute__((ext_vector_type(8)))  _Float16 v8h;
typedef __attribute__((ext_vector_type(8)))  float    v8f;

#define N_NODES  50000
#define N_EDGES  800000
#define F_IN     64
#define HID      256
#define N_GRAPHS 128
#define N_OUT    24
#define M_PAD    50048   // 782 * 64 (GEMM uses 64-row blocks)

// ---------------------------------------------------------------- utilities
__global__ void fill_zero_k(float* __restrict__ p, int n) {
    int i = blockIdx.x * blockDim.x + threadIdx.x;
    if (i < n) p[i] = 0.0f;
}

__global__ void deg_scatter_k(const long long* __restrict__ dst,
                              float* __restrict__ deg, int n_edges) {
    int e = blockIdx.x * blockDim.x + threadIdx.x;
    if (e < n_edges) atomicAdd(&deg[(int)dst[e]], 1.0f);
}

__global__ void deg_inv_k(float* __restrict__ deg, int n) {
    int i = blockIdx.x * blockDim.x + threadIdx.x;
    if (i < n) deg[i] = 1.0f / fmaxf(deg[i], 1.0f);
}

// agg[dst] += h[src]; one thread per (edge, 4-feature chunk)
__global__ void scatter_feat_k(const float* __restrict__ h,
                               const long long* __restrict__ src,
                               const long long* __restrict__ dst,
                               float* __restrict__ agg,
                               int n_edges, int F) {
    int chunks = F >> 2;
    int idx = blockIdx.x * blockDim.x + threadIdx.x;
    if (idx >= n_edges * chunks) return;
    int e = idx / chunks;
    int c = (idx - e * chunks) << 2;
    int s = (int)src[e];
    int d = (int)dst[e];
    const float4 v = *reinterpret_cast<const float4*>(h + (size_t)s * F + c);
    float* o = agg + (size_t)d * F + c;
    atomicAdd(o + 0, v.x);
    atomicAdd(o + 1, v.y);
    atomicAdd(o + 2, v.z);
    atomicAdd(o + 3, v.w);
}

// hcat[m, 0..F-1] = f16(agg[m]*deg_inv[m]); hcat[m, F..2F-1] = f16(h[m])
__global__ void build_hcat_k(const float* __restrict__ agg,
                             const float* __restrict__ h,
                             const float* __restrict__ deg_inv,
                             _Float16* __restrict__ hcat,
                             int M, int F) {
    int idx = blockIdx.x * blockDim.x + threadIdx.x;
    if (idx >= M * F) return;
    int m = idx / F;
    int f = idx - m * F;
    _Float16* row = hcat + (size_t)m * (2 * F);
    row[f]     = (_Float16)(agg[idx] * deg_inv[m]);
    row[F + f] = (_Float16)h[idx];
}

// wcat[n, 0..F-1] = f16(wl[n]); wcat[n, F..2F-1] = f16(wr[n])   (HID rows)
__global__ void build_wcat_k(const float* __restrict__ wl,
                             const float* __restrict__ wr,
                             _Float16* __restrict__ wcat, int F) {
    int idx = blockIdx.x * blockDim.x + threadIdx.x;
    if (idx >= HID * F) return;
    int n = idx / F;
    int f = idx - n * F;
    _Float16* row = wcat + (size_t)n * (2 * F);
    row[f]     = (_Float16)wl[idx];
    row[F + f] = (_Float16)wr[idx];
}

// ---------------------------------------------------------------- WMMA GEMM
// Out[M_PAD x 256] = relu( A[M_PAD x K] @ W[256 x K]^T + bias )
// Block = 256 threads = 8 waves. Block computes 64 rows x 256 cols.
// Wave w owns 32 output columns [w*32, w*32+32) and loops over 4 M-subtiles,
// reusing its 2 B fragments (registers) across all 4 -> 4x less B traffic.
// The 64x32-f16 A tile for each K-step is staged in LDS with CDNA5 async
// copies (GLOBAL_LOAD_ASYNC_TO_LDS_B128, ASYNCcnt), double-buffered so the
// kt+1 tile fetch overlaps the 8 WMMAs of step kt.
// f16 fragment layout per ISA: lane holds row (lane&15); half = lane>>4 holds
// K elements {k0..k0+7} u {k0+16..k0+23}, k0 = kt*32 + half*8.
template <int K>
__global__ __launch_bounds__(256) void sage_gemm_k(
    const _Float16* __restrict__ A,     // [M_PAD x K]   (hcat)
    const _Float16* __restrict__ W,     // [256 x K]     (wcat)
    const float* __restrict__ bias,     // [256]
    float* __restrict__ Out,            // [M_PAD x 256]
    int relu) {
    constexpr int N  = HID;
    constexpr int KT = K / 32;
    constexpr unsigned BUF_BYTES = 64 * 32 * 2;   // 4 KB per A-tile buffer

    __shared__ __align__(16) _Float16 atile[2 * 64 * 32];

    const int t    = threadIdx.x;
    const int lane = t & 31;
    const int wave = t >> 5;            // 0..7 -> 32-col group
    const int lm   = lane & 15;
    const int half = lane >> 4;
    const int row0 = blockIdx.x * 64;

    // async-copy mapping: each thread moves 16 bytes (8 f16) per K-step
    const int crow = t >> 2;            // 0..63   tile row
    const int ck   = (t & 3) * 8;       // 0,8,16,24  k element offset
    // LDS byte offset: flat LDS aperture keeps the offset in addr[31:0]
    const unsigned atile_off = (unsigned)(uintptr_t)(&atile[0]);
    const unsigned stage_lds = atile_off + (unsigned)(crow * 32 + ck) * 2u;
    const _Float16* stage_src = A + (size_t)(row0 + crow) * K + ck;

    v8f acc[4][2] = {};

    // ---- prologue: stage kt = 0 into buffer 0 ----
    {
        unsigned long long ga = (unsigned long long)(uintptr_t)stage_src;
        asm volatile("global_load_async_to_lds_b128 %0, %1, off"
                     :: "v"(stage_lds), "v"(ga) : "memory");
        asm volatile("s_wait_asynccnt 0x0" ::: "memory");
    }
    __syncthreads();

    int buf = 0;
#pragma unroll
    for (int kt = 0; kt < KT; ++kt) {
        // prefetch next A tile into the other buffer (overlaps the WMMAs)
        if (kt + 1 < KT) {
            unsigned dst = stage_lds + (unsigned)(buf ^ 1) * BUF_BYTES;
            unsigned long long ga =
                (unsigned long long)(uintptr_t)(stage_src + (kt + 1) * 32);
            asm volatile("global_load_async_to_lds_b128 %0, %1, off"
                         :: "v"(dst), "v"(ga) : "memory");
        }

        const int k0 = kt * 32 + half * 8;

        // B fragments for this wave's two 16-col subtiles (registers, global)
        v16h b[2];
#pragma unroll
        for (int j = 0; j < 2; ++j) {
            const _Float16* wr = W + (size_t)(wave * 32 + j * 16 + lm) * K;
            const v8h c0 = *reinterpret_cast<const v8h*>(wr + k0);
            const v8h c1 = *reinterpret_cast<const v8h*>(wr + k0 + 16);
#pragma unroll
            for (int i = 0; i < 8; ++i) { b[j][i] = c0[i]; b[j][8 + i] = c1[i]; }
        }

        // 4 M-subtiles from the LDS A tile, reusing b[0], b[1]
#pragma unroll
        for (int mt = 0; mt < 4; ++mt) {
            const _Float16* ab =
                &atile[buf * (64 * 32) + (mt * 16 + lm) * 32 + half * 8];
            const v8h c0 = *reinterpret_cast<const v8h*>(ab);
            const v8h c1 = *reinterpret_cast<const v8h*>(ab + 16);
            v16h a;
#pragma unroll
            for (int i = 0; i < 8; ++i) { a[i] = c0[i]; a[8 + i] = c1[i]; }
#pragma unroll
            for (int j = 0; j < 2; ++j) {
                acc[mt][j] = __builtin_amdgcn_wmma_f32_16x16x32_f16(
                    false, a, false, b[j], (short)0, acc[mt][j], false, false);
            }
        }

        if (kt + 1 < KT) {
            asm volatile("s_wait_asynccnt 0x0" ::: "memory");
            __syncthreads();
            buf ^= 1;
        }
    }

    // C/D layout: VGPR r -> M = r + 8*half; N = lane&15 (+ tile offsets)
#pragma unroll
    for (int mt = 0; mt < 4; ++mt) {
#pragma unroll
        for (int j = 0; j < 2; ++j) {
            const int n  = wave * 32 + j * 16 + lm;
            const float bn = bias[n];
#pragma unroll
            for (int r = 0; r < 8; ++r) {
                const int m = row0 + mt * 16 + half * 8 + r;
                float v = acc[mt][j][r] + bn;
                if (relu) v = fmaxf(v, 0.0f);
                Out[(size_t)m * N + n] = v;
            }
        }
    }
}

// ---------------------------------------------------------------- pooling/MLP
__global__ void pool_k(const float* __restrict__ h,
                       const long long* __restrict__ batch,
                       float* __restrict__ pooled,
                       float* __restrict__ counts, int M) {
    int idx = blockIdx.x * blockDim.x + threadIdx.x;
    if (idx >= M * HID) return;
    int m = idx >> 8;          // HID == 256
    int f = idx & 255;
    int g = (int)batch[m];
    atomicAdd(&pooled[g * HID + f], h[idx]);
    if (f == 0) atomicAdd(&counts[g], 1.0f);
}

__global__ void mlp1_k(const float* __restrict__ pooled,
                       const float* __restrict__ counts,
                       const float* __restrict__ w,   // [128 x 256]
                       const float* __restrict__ b,   // [128]
                       float* __restrict__ g1) {      // [128 x 128]
    int g = blockIdx.x;
    int o = threadIdx.x;
    float inv = 1.0f / fmaxf(counts[g], 1.0f);
    float s = 0.0f;
    for (int k = 0; k < HID; ++k) s += pooled[g * HID + k] * w[o * HID + k];
    g1[g * (HID / 2) + o] = fmaxf(s * inv + b[o], 0.0f);
}

__global__ void mlp2_k(const float* __restrict__ g1,  // [128 x 128]
                       const float* __restrict__ w,   // [24 x 128]
                       const float* __restrict__ b,   // [24]
                       float* __restrict__ out) {     // [128 x 24]
    int g = blockIdx.x;
    int o = threadIdx.x;
    if (o >= N_OUT) return;
    float s = 0.0f;
    for (int k = 0; k < HID / 2; ++k) s += g1[g * (HID / 2) + k] * w[o * (HID / 2) + k];
    out[g * N_OUT + o] = s + b[o];
}

// ---------------------------------------------------------------- launch
extern "C" void kernel_launch(void* const* d_in, const int* in_sizes, int n_in,
                              void* d_out, int out_size, void* d_ws, size_t ws_size,
                              hipStream_t stream) {
    (void)in_sizes; (void)n_in; (void)out_size; (void)ws_size;

    const float*      x     = (const float*)d_in[0];
    const long long*  edge  = (const long long*)d_in[1];
    const long long*  src   = edge;
    const long long*  dst   = edge + N_EDGES;
    const long long*  batch = (const long long*)d_in[2];
    const float* w1l = (const float*)d_in[3];
    const float* b1  = (const float*)d_in[4];
    const float* w1r = (const float*)d_in[5];
    const float* w2l = (const float*)d_in[6];
    const float* b2  = (const float*)d_in[7];
    const float* w2r = (const float*)d_in[8];
    const float* w3l = (const float*)d_in[9];
    const float* b3  = (const float*)d_in[10];
    const float* w3r = (const float*)d_in[11];
    const float* lin1_w = (const float*)d_in[12];
    const float* lin1_b = (const float*)d_in[13];
    const float* lin2_w = (const float*)d_in[14];
    const float* lin2_b = (const float*)d_in[15];
    float* out = (float*)d_out;

    // workspace carve-out (256B aligned slices)
    char* p = (char*)d_ws;
    auto take = [&](size_t bytes) -> char* {
        char* r = p;
        p += (bytes + 255) & ~(size_t)255;
        return r;
    };
    float*     deg    = (float*)take((size_t)N_NODES * 4);
    float*     agg    = (float*)take((size_t)N_NODES * HID * 4);
    float*     h      = (float*)take((size_t)M_PAD * HID * 4);
    _Float16*  hcat   = (_Float16*)take((size_t)M_PAD * 2 * HID * 2);
    _Float16*  wcat   = (_Float16*)take((size_t)HID * 2 * HID * 2);
    float*     pooled = (float*)take((size_t)N_GRAPHS * HID * 4);
    float*     counts = (float*)take((size_t)N_GRAPHS * 4);
    float*     g1     = (float*)take((size_t)N_GRAPHS * (HID / 2) * 4);

    auto nblk = [](long n, int b) { return dim3((unsigned)((n + b - 1) / b)); };

    // degree -> deg_inv (in place)
    fill_zero_k<<<nblk(N_NODES, 256), 256, 0, stream>>>(deg, N_NODES);
    deg_scatter_k<<<nblk(N_EDGES, 256), 256, 0, stream>>>(dst, deg, N_EDGES);
    deg_inv_k<<<nblk(N_NODES, 256), 256, 0, stream>>>(deg, N_NODES);

    // ---- layer 1 (F = 64, K = 128) ----
    {
        const int F = F_IN;
        fill_zero_k<<<nblk((long)N_NODES * F, 256), 256, 0, stream>>>(agg, N_NODES * F);
        scatter_feat_k<<<nblk((long)N_EDGES * (F / 4), 256), 256, 0, stream>>>(
            x, src, dst, agg, N_EDGES, F);
        build_hcat_k<<<nblk((long)N_NODES * F, 256), 256, 0, stream>>>(
            agg, x, deg, hcat, N_NODES, F);
        build_wcat_k<<<nblk((long)HID * F, 256), 256, 0, stream>>>(w1l, w1r, wcat, F);
        sage_gemm_k<2 * F_IN><<<dim3(M_PAD / 64), 256, 0, stream>>>(
            hcat, wcat, b1, h, 1);
    }

    // ---- layers 2 & 3 (F = 256, K = 512) ----
    const float* WL[2] = {w2l, w3l};
    const float* WR[2] = {w2r, w3r};
    const float* BB[2] = {b2, b3};
    for (int l = 0; l < 2; ++l) {
        const int F = HID;
        fill_zero_k<<<nblk((long)N_NODES * F, 256), 256, 0, stream>>>(agg, N_NODES * F);
        scatter_feat_k<<<nblk((long)N_EDGES * (F / 4), 256), 256, 0, stream>>>(
            h, src, dst, agg, N_EDGES, F);
        build_hcat_k<<<nblk((long)N_NODES * F, 256), 256, 0, stream>>>(
            agg, h, deg, hcat, N_NODES, F);
        build_wcat_k<<<nblk((long)HID * F, 256), 256, 0, stream>>>(WL[l], WR[l], wcat, F);
        sage_gemm_k<2 * HID><<<dim3(M_PAD / 64), 256, 0, stream>>>(
            hcat, wcat, BB[l], h, 1);  // safe: GEMM reads hcat (copy), writes h
    }

    // ---- global mean pool + MLP head ----
    fill_zero_k<<<nblk((long)N_GRAPHS * HID, 256), 256, 0, stream>>>(pooled, N_GRAPHS * HID);
    fill_zero_k<<<nblk(N_GRAPHS, 128), 128, 0, stream>>>(counts, N_GRAPHS);
    pool_k<<<nblk((long)N_NODES * HID, 256), 256, 0, stream>>>(h, batch, pooled, counts, N_NODES);
    mlp1_k<<<dim3(N_GRAPHS), dim3(HID / 2), 0, stream>>>(pooled, counts, lin1_w, lin1_b, g1);
    mlp2_k<<<dim3(N_GRAPHS), dim3(32), 0, stream>>>(g1, lin2_w, lin2_b, out);
}
